// TransGCN_22857815949367
// MI455X (gfx1250) — matrix operand order repaired and checked by reading, not compiled, not run on hardware
//
#include <hip/hip_runtime.h>
#include <hip/hip_bf16.h>

// MI455X (gfx1250): bf16 WMMA GEMMs (v_wmma_f32_16x16x32_bf16), LDS-staged
// fragments in ISA layout, fused MLP tile kernel, atomic scatter.
// Reference constants: F_IN = H = 128, FF = 2048, L = 2.

typedef __bf16 v16bf __attribute__((ext_vector_type(16)));
typedef float  v8f   __attribute__((ext_vector_type(8)));

#define H 128
#define FF_DIM 2048

__device__ __forceinline__ unsigned short f2bf(float f) {
  unsigned int u = __float_as_uint(f);
  unsigned int r = (u + 0x7FFFu + ((u >> 16) & 1u)) >> 16;   // round-to-nearest-even
  return (unsigned short)r;
}

__device__ __forceinline__ v16bf load_frag(const unsigned short* frag, int lane) {
  return *reinterpret_cast<const v16bf*>(frag + lane * 16);  // 32B aligned
}

__device__ __forceinline__ v8f wmma_bf16(v16bf a, v16bf b, v8f c) {
  return __builtin_amdgcn_wmma_f32_16x16x32_bf16(false, a, false, b, (short)0, c,
                                                 false, false);
}

// Stage a 128x32 A tile (rows m0..m0+127, cols k0..k0+31) from f32 (lda=128)
// into LDS in the ISA A-fragment layout (8 sub-tiles of 16x32, 512 ushorts each).
// A element (m,k): lane = (m%16) + 16*bit3(k), slot = (k&7) + 8*bit4(k).
__device__ __forceinline__ void stage_A_f32(const float* __restrict__ A, int nrows,
                                            int m0, int k0,
                                            unsigned short* __restrict__ frag, int tid) {
  int r  = tid >> 1;            // 0..127
  int kp = (tid & 1) * 16;      // 0 or 16
  int row = m0 + r;
  float vals[16];
  if (row < nrows) {
    const float4* p = (const float4*)(A + (size_t)row * H + k0 + kp);
#pragma unroll
    for (int i = 0; i < 4; ++i) {
      float4 v = p[i];
      vals[4*i+0] = v.x; vals[4*i+1] = v.y; vals[4*i+2] = v.z; vals[4*i+3] = v.w;
    }
  } else {
#pragma unroll
    for (int i = 0; i < 16; ++i) vals[i] = 0.0f;
  }
  unsigned short* base = frag + (r >> 4) * 512;
  int rr = r & 15;
#pragma unroll
  for (int i = 0; i < 16; ++i) {
    int k    = kp + i;
    int lane = rr + ((k >> 3) & 1) * 16;
    int j    = (k & 7) + ((k >> 4) & 1) * 8;
    base[lane * 16 + j] = f2bf(vals[i]);
  }
}

// Stage an nrows x 32 B tile from bf16 weight W [Nout x K] row-major
// (B = W^T) into ISA B-fragment layout: element (k,n): lane = (n%16)+16*bit3(k),
// slot = (k&7) + 8*bit4(k). One 16x32 sub-tile per 512 ushorts.
__device__ __forceinline__ void stage_B_bf16(const unsigned short* __restrict__ W, int ldw,
                                             int n0, int k0,
                                             unsigned short* __restrict__ frag,
                                             int tid, int nrows) {
  int r = tid >> 1;
  if (r >= nrows) return;
  int kp = (tid & 1) * 16;
  const unsigned short* src = W + (size_t)(n0 + r) * ldw + k0 + kp;
  unsigned short vals[16];
#pragma unroll
  for (int i = 0; i < 16; ++i) vals[i] = src[i];
  unsigned short* base = frag + (r >> 4) * 512;
  int nn = r & 15;
#pragma unroll
  for (int i = 0; i < 16; ++i) {
    int k = kp + i;
    base[(nn + ((k >> 3) & 1) * 16) * 16 + (k & 7) + ((k >> 4) & 1) * 8] = vals[i];
  }
}

// ---------------- GCN GEMM: C[M,128] = A[M,128] @ W[128,128]^T ----------------
// Full A and B tiles staged once (64 KB LDS), one barrier, then 32 WMMAs/wave.
__global__ __launch_bounds__(256) void gemm_nt_k128_kernel(
    const float* __restrict__ A, int M,
    const unsigned short* __restrict__ Wb,
    float* __restrict__ C) {
  __shared__ alignas(32) unsigned short sA[16384];  // 4 ksteps * 8 mt * 512
  __shared__ alignas(32) unsigned short sB[16384];  // 4 ksteps * 8 nt * 512
  int tid = threadIdx.x, lane = tid & 31, w = tid >> 5;
  int m0 = blockIdx.x * 128;

#pragma unroll
  for (int ks = 0; ks < 4; ++ks) {
    stage_A_f32(A, M, m0, ks * 32, sA + ks * 4096, tid);
    stage_B_bf16(Wb, H, 0, ks * 32, sB + ks * 4096, tid, 128);
  }
  __syncthreads();

  v8f acc[8];
#pragma unroll
  for (int i = 0; i < 8; ++i) acc[i] = (v8f)(0.0f);

#pragma unroll
  for (int ks = 0; ks < 4; ++ks) {
    const unsigned short* aB = sA + ks * 4096;
    v16bf b = load_frag(sB + ks * 4096 + w * 512, lane);
    // grouped preload (4 frags) -> 4 WMMAs: lets the compiler batch ds waits
    v16bf a0 = load_frag(aB + 0 * 512, lane);
    v16bf a1 = load_frag(aB + 1 * 512, lane);
    v16bf a2 = load_frag(aB + 2 * 512, lane);
    v16bf a3 = load_frag(aB + 3 * 512, lane);
    acc[0] = wmma_bf16(a0, b, acc[0]);
    acc[1] = wmma_bf16(a1, b, acc[1]);
    acc[2] = wmma_bf16(a2, b, acc[2]);
    acc[3] = wmma_bf16(a3, b, acc[3]);
    v16bf a4 = load_frag(aB + 4 * 512, lane);
    v16bf a5 = load_frag(aB + 5 * 512, lane);
    v16bf a6 = load_frag(aB + 6 * 512, lane);
    v16bf a7 = load_frag(aB + 7 * 512, lane);
    acc[4] = wmma_bf16(a4, b, acc[4]);
    acc[5] = wmma_bf16(a5, b, acc[5]);
    acc[6] = wmma_bf16(a6, b, acc[6]);
    acc[7] = wmma_bf16(a7, b, acc[7]);
  }

  int col = w * 16 + (lane & 15);
  int rb  = m0 + ((lane >> 4) << 3);
#pragma unroll
  for (int mt = 0; mt < 8; ++mt)
#pragma unroll
    for (int v = 0; v < 8; ++v) {
      int row = rb + mt * 16 + v;
      if (row < M) C[(size_t)row * H + col] = acc[mt][v];
    }
}

// ------------- Fused MLP tile: out = (act(t@W1^T+b1))@W2^T + b2 + res --------
// W1: [F,128] bf16, W2: [128,F] bf16. 128-row tile, F chunked by 64; the
// intermediate never touches HBM.
template <int RELU1>
__global__ __launch_bounds__(256) void fused_mlp_kernel(
    const float* __restrict__ Tin, int M,
    const unsigned short* __restrict__ W1b, const float* __restrict__ b1,
    const unsigned short* __restrict__ W2b, const float* __restrict__ b2,
    const float* __restrict__ res, float* __restrict__ out, int F) {
  __shared__ alignas(32) unsigned short sT[16384];  // 4 ksteps * 8 mt * 512
  __shared__ alignas(32) unsigned short sWB[8192];  // W1 chunk / W2 chunk (union)
  __shared__ alignas(32) unsigned short sU[8192];   // intermediate, A-frag order
  int tid = threadIdx.x, lane = tid & 31, w = tid >> 5;
  int m0 = blockIdx.x * 128;

  // Stage the whole 128x128 t tile once (f32 -> bf16, A-fragment order).
#pragma unroll
  for (int ks = 0; ks < 4; ++ks)
    stage_A_f32(Tin, M, m0, ks * 32, sT + ks * 4096, tid);

  v8f cacc[8];
#pragma unroll
  for (int i = 0; i < 8; ++i) cacc[i] = (v8f)(0.0f);

  int mtb = (w >> 2) << 2;   // 0 or 4: this wave's mt group in GEMM1
  int chunks = F >> 6;
  for (int c = 0; c < chunks; ++c) {
    __syncthreads();  // prev chunk done reading sWB/sU (publishes sT at c==0)
    // ---- stage W1 chunk: rows [c*64, c*64+64), K = 128
#pragma unroll
    for (int ks = 0; ks < 4; ++ks)
      stage_B_bf16(W1b, H, c * 64, ks * 32, sWB + ks * 2048, tid, 64);
    __syncthreads();

    // ---- GEMM1: U[128x64] = T @ W1c^T ; wave w: nt = w&3, mts = mtb..mtb+3
    v8f uacc[4];
#pragma unroll
    for (int i = 0; i < 4; ++i) uacc[i] = (v8f)(0.0f);
#pragma unroll
    for (int ks = 0; ks < 4; ++ks) {
      v16bf b  = load_frag(sWB + ks * 2048 + (w & 3) * 512, lane);
      v16bf a0 = load_frag(sT + ks * 4096 + (mtb + 0) * 512, lane);
      v16bf a1 = load_frag(sT + ks * 4096 + (mtb + 1) * 512, lane);
      v16bf a2 = load_frag(sT + ks * 4096 + (mtb + 2) * 512, lane);
      v16bf a3 = load_frag(sT + ks * 4096 + (mtb + 3) * 512, lane);
      uacc[0] = wmma_bf16(a0, b, uacc[0]);
      uacc[1] = wmma_bf16(a1, b, uacc[1]);
      uacc[2] = wmma_bf16(a2, b, uacc[2]);
      uacc[3] = wmma_bf16(a3, b, uacc[3]);
    }
    __syncthreads();  // all waves done reading sWB (W1)

    // ---- bias + (relu) + write U to LDS in A-fragment order (K dim = 64)
    int nloc = (w & 3) * 16 + (lane & 15);      // local ff column 0..63
    float b1v = b1[c * 64 + nloc];
#pragma unroll
    for (int i = 0; i < 4; ++i) {
      int mt = mtb + i;
#pragma unroll
      for (int v = 0; v < 8; ++v) {
        int row = mt * 16 + v + ((lane >> 4) << 3);
        float val = uacc[i][v] + b1v;
        if (RELU1) val = fmaxf(val, 0.0f);
        int kk    = nloc & 31;
        int lane2 = (row & 15) + ((kk >> 3) & 1) * 16;
        int j     = (kk & 7) + ((kk >> 4) & 1) * 8;
        sU[(nloc >> 5) * 4096 + (row >> 4) * 512 + lane2 * 16 + j] = f2bf(val);
      }
    }
    // ---- stage W2 chunk (overwrites W1 region): 128 rows, K cols c*64..+64
#pragma unroll
    for (int ks2 = 0; ks2 < 2; ++ks2)
      stage_B_bf16(W2b, F, 0, c * 64 + ks2 * 32, sWB + ks2 * 4096, tid, 128);
    __syncthreads();

    // ---- GEMM2 accumulate: C += U @ W2c^T ; wave w owns nt = w
#pragma unroll
    for (int ks2 = 0; ks2 < 2; ++ks2) {
      v16bf b  = load_frag(sWB + ks2 * 4096 + w * 512, lane);
      const unsigned short* uB = sU + ks2 * 4096;
      v16bf a0 = load_frag(uB + 0 * 512, lane);
      v16bf a1 = load_frag(uB + 1 * 512, lane);
      v16bf a2 = load_frag(uB + 2 * 512, lane);
      v16bf a3 = load_frag(uB + 3 * 512, lane);
      cacc[0] = wmma_bf16(a0, b, cacc[0]);
      cacc[1] = wmma_bf16(a1, b, cacc[1]);
      cacc[2] = wmma_bf16(a2, b, cacc[2]);
      cacc[3] = wmma_bf16(a3, b, cacc[3]);
      v16bf a4 = load_frag(uB + 4 * 512, lane);
      v16bf a5 = load_frag(uB + 5 * 512, lane);
      v16bf a6 = load_frag(uB + 6 * 512, lane);
      v16bf a7 = load_frag(uB + 7 * 512, lane);
      cacc[4] = wmma_bf16(a4, b, cacc[4]);
      cacc[5] = wmma_bf16(a5, b, cacc[5]);
      cacc[6] = wmma_bf16(a6, b, cacc[6]);
      cacc[7] = wmma_bf16(a7, b, cacc[7]);
    }
  }

  // ---- epilogue: + b2 + residual
  int col = w * 16 + (lane & 15);
  float b2v = b2[col];
  int rb = m0 + ((lane >> 4) << 3);
#pragma unroll
  for (int mt = 0; mt < 8; ++mt)
#pragma unroll
    for (int v = 0; v < 8; ++v) {
      int row = rb + mt * 16 + v;
      if (row < M) {
        size_t idx = (size_t)row * H + col;
        out[idx] = cacc[mt][v] + b2v + res[idx];
      }
    }
}

// ---------------------------- small helper kernels ---------------------------
__global__ void fill0_kernel(float* p, int n) {
  int i = blockIdx.x * 256 + threadIdx.x;
  if (i < n) p[i] = 0.0f;
}

__global__ void f2bf_kernel(const float* __restrict__ src, unsigned short* __restrict__ dst, int n) {
  int i = blockIdx.x * 256 + threadIdx.x;
  if (i < n) dst[i] = f2bf(src[i]);
}

__global__ void deg_kernel(const int* __restrict__ col, float* __restrict__ deg, int E) {
  int e = blockIdx.x * 256 + threadIdx.x;
  if (e < E) atomicAdd(&deg[col[e]], 1.0f);
}

__global__ void dinv_kernel(const float* __restrict__ deg, float* __restrict__ dinv, int n) {
  int i = blockIdx.x * 256 + threadIdx.x;
  if (i < n) dinv[i] = rsqrtf(deg[i] + 1.0f);   // +1 for self-loop
}

__global__ void scatter_kernel(const int* __restrict__ row, const int* __restrict__ col,
                               const float* __restrict__ dinv, const float* __restrict__ h,
                               float* __restrict__ agg, int E) {
  int t = blockIdx.x * 256 + threadIdx.x;
  int e = t >> 5;
  if (e >= E) return;
  int f = (t & 31) << 2;
  int r = row[e], c = col[e];
  float nrm = dinv[r] * dinv[c];
  float4 hv = *(const float4*)(h + (size_t)r * H + f);
  float* dst = agg + (size_t)c * H + f;
  atomicAdd(dst + 0, nrm * hv.x);
  atomicAdd(dst + 1, nrm * hv.y);
  atomicAdd(dst + 2, nrm * hv.z);
  atomicAdd(dst + 3, nrm * hv.w);
}

__global__ void gcn_finish_kernel(const float* __restrict__ agg, const float* __restrict__ hpre,
                                  const float* __restrict__ dinv, const float* __restrict__ bias,
                                  float* __restrict__ out, int N) {
  int t = blockIdx.x * 256 + threadIdx.x;
  int r = t >> 5;
  if (r >= N) return;
  int f = (t & 31) << 2;
  float di = dinv[r];
  float d2 = di * di;
  size_t idx = (size_t)r * H + f;
  float4 a = *(const float4*)(agg + idx);
  float4 h = *(const float4*)(hpre + idx);
  float4 b = *(const float4*)(bias + f);
  float4 o;
  o.x = fmaxf(a.x + d2 * h.x + b.x, 0.0f);
  o.y = fmaxf(a.y + d2 * h.y + b.y, 0.0f);
  o.z = fmaxf(a.z + d2 * h.z + b.z, 0.0f);
  o.w = fmaxf(a.w + d2 * h.w + b.w, 0.0f);
  *(float4*)(out + idx) = o;
}

__global__ __launch_bounds__(256) void layernorm_kernel(
    const float* __restrict__ x, const float* __restrict__ g, const float* __restrict__ b,
    float* __restrict__ out, int M) {
  int wave = threadIdx.x >> 5, lane = threadIdx.x & 31;
  int row = blockIdx.x * 8 + wave;
  if (row >= M) return;
  size_t base = (size_t)row * H;
  float4 v = *(const float4*)(x + base + lane * 4);
  float s = v.x + v.y + v.z + v.w;
#pragma unroll
  for (int off = 16; off > 0; off >>= 1) s += __shfl_xor(s, off, 32);
  float mean = s * (1.0f / 128.0f);
  float dx = v.x - mean, dy = v.y - mean, dz = v.z - mean, dw = v.w - mean;
  float q = dx * dx + dy * dy + dz * dz + dw * dw;
#pragma unroll
  for (int off = 16; off > 0; off >>= 1) q += __shfl_xor(q, off, 32);
  float rstd = rsqrtf(q * (1.0f / 128.0f) + 1e-5f);
  float4 gg = *(const float4*)(g + lane * 4);
  float4 bb = *(const float4*)(b + lane * 4);
  float4 o;
  o.x = dx * rstd * gg.x + bb.x;
  o.y = dy * rstd * gg.y + bb.y;
  o.z = dz * rstd * gg.z + bb.z;
  o.w = dw * rstd * gg.w + bb.w;
  *(float4*)(out + base + lane * 4) = o;
}

__global__ void final_kernel(const float* __restrict__ t, const float* __restrict__ hg,
                             float* __restrict__ out, int n4) {
  int i = blockIdx.x * 256 + threadIdx.x;
  if (i >= n4) return;
  float4 a = ((const float4*)t)[i];
  float4 b = ((const float4*)hg)[i];
  float4 o;
  o.x = fmaxf(a.x + b.x, 0.0f);
  o.y = fmaxf(a.y + b.y, 0.0f);
  o.z = fmaxf(a.z + b.z, 0.0f);
  o.w = fmaxf(a.w + b.w, 0.0f);
  ((float4*)out)[i] = o;
}

// --------------------------------- launcher ----------------------------------
extern "C" void kernel_launch(void* const* d_in, const int* in_sizes, int n_in,
                              void* d_out, int out_size, void* d_ws, size_t ws_size,
                              hipStream_t stream) {
  const float* x     = (const float*)d_in[0];
  const int*   ei    = (const int*)d_in[1];
  const float* gcn_w = (const float*)d_in[2];
  const float* gcn_b = (const float*)d_in[3];
  const float* in_w  = (const float*)d_in[4];   // [2, 384, 128]
  const float* in_b  = (const float*)d_in[5];   // [2, 384]
  const float* out_w = (const float*)d_in[6];   // [2, 128, 128]
  const float* out_b = (const float*)d_in[7];   // [2, 128]
  const float* ln1_g = (const float*)d_in[8];
  const float* ln1_b = (const float*)d_in[9];
  const float* ff1_w = (const float*)d_in[10];  // [2, 2048, 128]
  const float* ff1_b = (const float*)d_in[11];  // [2, 2048]
  const float* ff2_w = (const float*)d_in[12];  // [2, 128, 2048]
  const float* ff2_b = (const float*)d_in[13];  // [2, 128]
  const float* ln2_g = (const float*)d_in[14];
  const float* ln2_b = (const float*)d_in[15];
  float* out = (float*)d_out;

  const int N = in_sizes[0] / H;     // 50000
  const int E = in_sizes[1] / 2;     // 800000
  const size_t NH = (size_t)N * H;

  // ---- workspace carve-up (assumes ws_size >= ~110 MB) ----
  char* wp = (char*)d_ws;
  auto alloc = [&](size_t bytes) -> void* {
    void* p = (void*)wp;
    wp += (bytes + 255) & ~(size_t)255;
    return p;
  };
  float* deg  = (float*)alloc((size_t)N * 4);
  float* dinv = (float*)alloc((size_t)N * 4);
  float* buf0 = (float*)alloc(NH * 4);   // h_pre, later pre-LN buffer
  float* buf1 = (float*)alloc(NH * 4);   // agg, later t_A
  float* hg   = (float*)alloc(NH * 4);   // GCN output (residual for encoder)
  float* tB   = (float*)alloc(NH * 4);   // t after each full layer
  unsigned short* gcnwb = (unsigned short*)alloc((size_t)H * H * 2);
  unsigned short* wvb[2], *wob[2], *ff1b[2], *ff2b[2];
  for (int l = 0; l < 2; ++l) {
    wvb[l]  = (unsigned short*)alloc((size_t)H * H * 2);
    wob[l]  = (unsigned short*)alloc((size_t)H * H * 2);
    ff1b[l] = (unsigned short*)alloc((size_t)FF_DIM * H * 2);
    ff2b[l] = (unsigned short*)alloc((size_t)H * FF_DIM * 2);
  }

  dim3 blk(256);
  auto g1 = [](size_t n) { return dim3((unsigned)((n + 255) / 256)); };

  // zero deg + agg
  fill0_kernel<<<g1(N), blk, 0, stream>>>(deg, N);
  fill0_kernel<<<g1(NH), blk, 0, stream>>>(buf1, (int)NH);

  // degree / norm
  deg_kernel<<<g1(E), blk, 0, stream>>>(ei + E, deg, E);
  dinv_kernel<<<g1(N), blk, 0, stream>>>(deg, dinv, N);

  // convert weights to bf16
  f2bf_kernel<<<g1(H * H), blk, 0, stream>>>(gcn_w, gcnwb, H * H);
  for (int l = 0; l < 2; ++l) {
    f2bf_kernel<<<g1(H * H), blk, 0, stream>>>(in_w + (size_t)l * 384 * H + 256 * H, wvb[l], H * H);
    f2bf_kernel<<<g1(H * H), blk, 0, stream>>>(out_w + (size_t)l * H * H, wob[l], H * H);
    f2bf_kernel<<<g1(FF_DIM * H), blk, 0, stream>>>(ff1_w + (size_t)l * FF_DIM * H, ff1b[l], FF_DIM * H);
    f2bf_kernel<<<g1(FF_DIM * H), blk, 0, stream>>>(ff2_w + (size_t)l * FF_DIM * H, ff2b[l], FF_DIM * H);
  }

  dim3 gM((N + 127) / 128);
  dim3 gLN((N + 7) / 8);

  // GCN: h_pre = x @ gcn_w^T
  gemm_nt_k128_kernel<<<gM, blk, 0, stream>>>(x, N, gcnwb, buf0);
  // agg[col] += dinv[row]*dinv[col] * h_pre[row]
  scatter_kernel<<<g1((size_t)E * 32), blk, 0, stream>>>(ei, ei + E, dinv, buf0, buf1, E);
  // hg = relu(agg + dinv^2 * h_pre + b)
  gcn_finish_kernel<<<g1((size_t)N * 32), blk, 0, stream>>>(buf1, buf0, dinv, gcn_b, hg, N);

  // transformer encoder (seq=1 -> MHA == out_proj(v_proj(x))), post-LN
  const float* t_in = hg;
  for (int l = 0; l < 2; ++l) {
    const float* bv = in_b + (size_t)l * 384 + 256;
    const float* bo = out_b + (size_t)l * H;
    // pre1 = t + out_proj(v_proj(t))
    fused_mlp_kernel<0><<<gM, blk, 0, stream>>>(t_in, N, wvb[l], bv, wob[l], bo, t_in, buf0, H);
    layernorm_kernel<<<gLN, blk, 0, stream>>>(buf0, ln1_g + (size_t)l * H, ln1_b + (size_t)l * H, buf1, N);
    // pre2 = tA + ff2(relu(ff1(tA)))
    fused_mlp_kernel<1><<<gM, blk, 0, stream>>>(buf1, N, ff1b[l], ff1_b + (size_t)l * FF_DIM,
                                                ff2b[l], ff2_b + (size_t)l * H, buf1, buf0, FF_DIM);
    layernorm_kernel<<<gLN, blk, 0, stream>>>(buf0, ln2_g + (size_t)l * H, ln2_b + (size_t)l * H, tB, N);
    t_in = tB;
  }

  // out = relu(t + hg)
  final_kernel<<<g1(NH / 4), blk, 0, stream>>>(tB, hg, out, (int)(NH / 4));
}
